// GNNAutoEncoder_893353198359
// MI455X (gfx1250) — matrix operand order repaired and checked by reading, not compiled, not run on hardware
//
#include <hip/hip_runtime.h>

typedef __attribute__((ext_vector_type(16))) __bf16 v16bf;
typedef __attribute__((ext_vector_type(8)))  __bf16 v8bf;
typedef __attribute__((ext_vector_type(8)))  float  v8f;

// Round-to-nearest-even f32 -> bf16 bits, branch-free, no libcalls.
static __device__ __forceinline__ unsigned f2bfbits(float f) {
    unsigned u = __builtin_bit_cast(unsigned, f);
    u += 0x7FFFu + ((u >> 16) & 1u);
    return u >> 16;
}

__global__ void k_zero(float* __restrict__ p, int n) {
    int i = blockIdx.x * blockDim.x + threadIdx.x;
    if (i < n) p[i] = 0.0f;
}

__global__ void k_deg(const long long* __restrict__ dst, float* __restrict__ deg, int E) {
    int e = blockIdx.x * blockDim.x + threadIdx.x;
    if (e < E) unsafeAtomicAdd(&deg[dst[e]], 1.0f);
}

__global__ void k_rsqrt(float* __restrict__ deg, int n) {
    int i = blockIdx.x * blockDim.x + threadIdx.x;
    if (i < n) deg[i] = rsqrtf(deg[i] + 1.0f);  // self-loop included
}

// Convert W[K x Nc] (f32 row-major) -> WT[Nc x K] (bf16): per-lane contiguous
// K-runs for the WMMA B fragment.
__global__ void k_wprep(const float* __restrict__ W, unsigned short* __restrict__ WT,
                        int K, int Nc) {
    int i = blockIdx.x * blockDim.x + threadIdx.x;
    if (i < K * Nc) {
        int k = i / Nc, n = i - k * Nc;
        WT[(size_t)n * K + k] = (unsigned short)f2bfbits(W[i]);
    }
}

// Packed f32 -> bf16 conversion: 2 elements per thread, one dword store.
__global__ void k_pack2bf(const float* __restrict__ in, unsigned* __restrict__ out,
                          int pairs) {
    int i = blockIdx.x * blockDim.x + threadIdx.x;
    if (i < pairs) {
        float2 v = ((const float2*)in)[i];
        out[i] = f2bfbits(v.x) | (f2bfbits(v.y) << 16);
    }
}

// C[M x Nc] = A[M x K] @ B  (+bias).
// A: bf16 row-major. B: pre-converted bf16 column-major WT[Nc x K].
// One wave32 computes one 16x16 tile via V_WMMA_F32_16X16X32_BF16.
// Requires M % 16 == 0 (true: 100000). Block: (Nc/16) waves. Grid.x: M/16.
template <int K>
__global__ __launch_bounds__(256) void k_gemm_bf16(
    const unsigned short* __restrict__ A, const unsigned short* __restrict__ WT,
    const float* __restrict__ bias, float* __restrict__ C, int Nc) {
    const int lane = threadIdx.x & 31;
    const int wave = threadIdx.x >> 5;
    const int n0   = wave * 16;
    const int m0   = blockIdx.x * 16;
    const int l15  = lane & 15;
    const int hs   = lane >> 4;   // 0 => lanes 0-15, 1 => lanes 16-31

    const unsigned short* arow = A  + (size_t)(m0 + l15) * K;
    const int             col  = n0 + l15;
    const unsigned short* brow = WT + (size_t)col * K;

    v8f acc = {};
    #pragma unroll
    for (int k0 = 0; k0 < K; k0 += 32) {
        // A fragment: lanes 0-15 hold K = k0..k0+15,
        //             lanes 16-31 hold K = k0+16..k0+31.
        v16bf a = *(const v16bf*)(arow + k0 + hs * 16);

        // B fragment: lane = column col; lanes 0-15 K=k0..k0+15,
        // lanes 16-31 K=k0+16..k0+31 -> one contiguous 32B bf16 load.
        v16bf b = *(const v16bf*)(brow + k0 + hs * 16);

        acc = __builtin_amdgcn_wmma_f32_16x16x32_bf16(
            /*neg_a=*/false, a, /*neg_b=*/false, b,
            /*c_mod=*/(short)0, acc, /*reuse_a=*/false, /*reuse_b=*/false);
    }

    const float bcol = bias ? bias[col] : 0.0f;
    // D layout: VGPR r -> row m0 + hs*8 + r, column col.
    #pragma unroll
    for (int r = 0; r < 8; ++r)
        C[(size_t)(m0 + hs * 8 + r) * Nc + col] = acc[r] + bcol;
}

// agg[n][c] = h[n][c] * dinv[n]^2   (self-loop term; C fixed at 64)
__global__ void k_self64(const float* __restrict__ h, const float* __restrict__ dinv,
                         float* __restrict__ agg, int total) {
    int i = blockIdx.x * blockDim.x + threadIdx.x;
    if (i < total) {
        float d = dinv[i >> 6];
        agg[i] = h[i] * d * d;
    }
}

// agg[dst] += h[src] * dinv[src] * dinv[dst]; 64 lanes per edge (coalesced),
// L2-resident hardware f32 atomics.
__global__ __launch_bounds__(256) void k_scatter64(
    const float* __restrict__ h, const long long* __restrict__ src,
    const long long* __restrict__ dst, const float* __restrict__ dinv,
    float* __restrict__ agg, int E) {
    int e = blockIdx.x * 4 + (threadIdx.x >> 6);
    int c = threadIdx.x & 63;
    if (e < E) {
        long long s = src[e], d = dst[e];
        float norm = dinv[s] * dinv[d];
        float v = h[(size_t)s * 64 + c] * norm;
        unsafeAtomicAdd(&agg[(size_t)d * 64 + c], v);
    }
}

// out_bf16[2i,2i+1] = pack(act(agg + bias)); C fixed at 64, 2 elems/thread.
__global__ void k_biasbf64(const float* __restrict__ agg, const float* __restrict__ b,
                           unsigned* __restrict__ outbf, int pairs, int relu) {
    int i = blockIdx.x * blockDim.x + threadIdx.x;
    if (i < pairs) {
        int idx = i * 2;
        int c = idx & 63;
        float2 v = ((const float2*)agg)[i];
        float v0 = v.x + b[c];
        float v1 = v.y + b[c + 1];
        if (relu) { v0 = fmaxf(v0, 0.0f); v1 = fmaxf(v1, 0.0f); }
        outbf[i] = f2bfbits(v0) | (f2bfbits(v1) << 16);
    }
}

extern "C" void kernel_launch(void* const* d_in, const int* in_sizes, int n_in,
                              void* d_out, int out_size, void* d_ws, size_t ws_size,
                              hipStream_t stream) {
    const float*     x  = (const float*)d_in[0];
    const long long* ei = (const long long*)d_in[1];   // int64 edge_index [2, E]
    const float*     W1 = (const float*)d_in[2];
    const float*     b1 = (const float*)d_in[3];
    const float*     W2 = (const float*)d_in[4];
    const float*     b2 = (const float*)d_in[5];
    const float*     Wd = (const float*)d_in[6];
    const float*     bd = (const float*)d_in[7];
    float*           out = (float*)d_out;

    const int N = in_sizes[0] / 128;   // 100000 (divisible by 16)
    const int E = in_sizes[1] / 2;     // 1600000
    const long long* src = ei;
    const long long* dst = ei + E;

    // Workspace (f32 units): dinv[npad] | bf16 weights (10240) | bufA[N*64]
    //                        | bufB[N*64] | actbf[N*64]  (~77 MB total)
    float* ws   = (float*)d_ws;
    size_t npad = ((size_t)N + 127) & ~(size_t)127;
    float* dinv = ws;
    unsigned short* wt1 = (unsigned short*)(ws + npad);   // 128*64
    unsigned short* wt2 = wt1 + 128 * 64;                 // 64*64
    unsigned short* wtd = wt2 + 64 * 64;                  // 64*128
    float* bufA = ws + npad + 10240;                      // pre-agg GEMM out (f32)
    float* bufB = bufA + (size_t)N * 64;                  // aggregation target (f32)
    // bf16 activations: x (N*128 bf16) first, then reused for h1/z (N*64 bf16)
    unsigned short* actbf = (unsigned short*)(bufB + (size_t)N * 64);

    const int total = N * 64;
    const int gN  = (N + 255) / 256;
    const int gE  = (E + 255) / 256;
    const int gT  = (total + 255) / 256;
    const int gP  = (total / 2 + 255) / 256;   // pair-granular kernels over N*64
    const int gSc = (E + 3) / 4;
    const int gM  = N / 16;

    // --- weight prep (tiny) + degree/normalization + x -> bf16 ---
    k_wprep<<<(128 * 64 + 255) / 256, 256, 0, stream>>>(W1, wt1, 128, 64);
    k_wprep<<<(64 * 64 + 255) / 256, 256, 0, stream>>>(W2, wt2, 64, 64);
    k_wprep<<<(64 * 128 + 255) / 256, 256, 0, stream>>>(Wd, wtd, 64, 128);
    k_zero <<<gN, 256, 0, stream>>>(dinv, N);
    k_deg  <<<gE, 256, 0, stream>>>(dst, dinv, E);
    k_rsqrt<<<gN, 256, 0, stream>>>(dinv, N);
    k_pack2bf<<<gT, 256, 0, stream>>>(x, (unsigned*)actbf, N * 64); // N*128/2 pairs

    // --- layer 1: h1 = relu(Agg(x@W1) + b1) ---
    k_gemm_bf16<128><<<gM, 128, 0, stream>>>(actbf, wt1, nullptr, bufA, 64);
    k_self64   <<<gT, 256, 0, stream>>>(bufA, dinv, bufB, total);
    k_scatter64<<<gSc, 256, 0, stream>>>(bufA, src, dst, dinv, bufB, E);
    k_biasbf64 <<<gP, 256, 0, stream>>>(bufB, b1, (unsigned*)actbf, total / 2, 1);

    // --- layer 2: z = Agg(h1@W2) + b2 ---
    k_gemm_bf16<64><<<gM, 128, 0, stream>>>(actbf, wt2, nullptr, bufA, 64);
    k_self64   <<<gT, 256, 0, stream>>>(bufA, dinv, bufB, total);
    k_scatter64<<<gSc, 256, 0, stream>>>(bufA, src, dst, dinv, bufB, E);
    k_biasbf64 <<<gP, 256, 0, stream>>>(bufB, b2, (unsigned*)actbf, total / 2, 0);

    // --- decoder: out = z @ Wd + bd ---
    k_gemm_bf16<64><<<gM, 256, 0, stream>>>(actbf, wtd, bd, out, 128);
}
